// AttentionBlock_34565896798661
// MI455X (gfx1250) — matrix operand order repaired and checked by reading, not compiled, not run on hardware
//
#include <hip/hip_runtime.h>

typedef __bf16 bf16_t;
typedef __attribute__((ext_vector_type(16))) __bf16 v16bf;
typedef __attribute__((ext_vector_type(8)))  float  v8f;
typedef __attribute__((ext_vector_type(4)))  int    v4i;

#define HEADS 8
#define BATCH 4
#define SEQ   2048
#define INSZ  1024
#define KD    64
#define VD    128

#if defined(__AMDGCN__) && __has_builtin(__builtin_amdgcn_global_load_async_to_lds_b128)
#define HAVE_ASYNC_LDS 1
#else
#define HAVE_ASYNC_LDS 0
#endif

__device__ __forceinline__ void wait_asynccnt0() {
#if defined(__AMDGCN__)
#if __has_builtin(__builtin_amdgcn_s_wait_asynccnt)
  __builtin_amdgcn_s_wait_asynccnt(0);
#else
  asm volatile("s_wait_asynccnt 0" ::: "memory");
#endif
#endif
}

__device__ __forceinline__ v8f wmma_bf16(v16bf a, v16bf b, v8f c) {
  // D = A(16x32 bf16) * B(32x16 bf16) + C(16x16 f32)
  return __builtin_amdgcn_wmma_f32_16x16x32_bf16(
      /*neg_a=*/false, a, /*neg_b=*/false, b,
      /*c_mod=*/(short)0, c, /*reuse_a=*/false, /*reuse_b=*/false);
}

// A fragment: 16x32 bf16 tile at `src` (row-major, leading dim ld).
// Lane l holds row m = l&15; VGPR j holds K = (j<4?0:16) + (l>>4)*8 + (j%4)*2 .. +1
__device__ __forceinline__ v16bf load_a_frag(const bf16_t* src, int ld, int lane) {
  int m = lane & 15, half = lane >> 4;
  const bf16_t* p = src + (size_t)m * ld;
  v16bf a;
#pragma unroll
  for (int j = 0; j < 8; ++j) {
    int kb = ((j & 4) ? 16 : 0) + half * 8 + (j & 3) * 2;
    a[2 * j]     = p[kb];
    a[2 * j + 1] = p[kb + 1];
  }
  return a;
}

// B fragment from B-transposed storage: bt[n][k] row-major with leading dim ld.
// b[2j] = B[K][n] = bt[n][K], K = (j<4?0:16) + (lane>>4)*8 + (j%4)*2, n = lane&15.
__device__ __forceinline__ v16bf load_bT_frag(const bf16_t* bt, int ld, int lane) {
  int n = lane & 15, koff = (lane >> 4) * 8;
  const bf16_t* p = bt + (size_t)n * ld + koff;
  v16bf b;
#pragma unroll
  for (int j = 0; j < 8; ++j) {
    b[2 * j]     = p[((j & 4) ? 16 : 0) + (j & 3) * 2];
    b[2 * j + 1] = p[((j & 4) ? 16 : 0) + (j & 3) * 2 + 1];
  }
  return b;
}

__global__ __launch_bounds__(256) void cvt_f32_bf16(const float* __restrict__ in,
                                                    bf16_t* __restrict__ out, int n) {
  for (int i = blockIdx.x * blockDim.x + threadIdx.x; i < n; i += gridDim.x * blockDim.x)
    out[i] = (bf16_t)in[i];
}

// M: [H][1][INSZ][D] f32 -> Wt: [H][D][INSZ] bf16 (transposed per head)
__global__ __launch_bounds__(256) void pack_weight_t(const float* __restrict__ M,
                                                     bf16_t* __restrict__ Wt, int D) {
  int total = HEADS * INSZ * D;
  for (int i = blockIdx.x * blockDim.x + threadIdx.x; i < total; i += gridDim.x * blockDim.x) {
    int h   = i / (INSZ * D);
    int rem = i - h * INSZ * D;
    int ii  = rem / D;
    int d   = rem - ii * D;
    Wt[((size_t)h * D + d) * INSZ + ii] = (bf16_t)M[i];
  }
}

// out[h][gr][d] (gr = b*SEQ+s) = sum_i xb[gr][i] * Wt[h][d][i]
// grid: (8192/128, HEADS, D/64); block 256 = 8 waves; wave w: rows gr0+w*16, 4 N-tiles.
__global__ __launch_bounds__(256) void proj_gemm(const bf16_t* __restrict__ xb,
                                                 const bf16_t* __restrict__ Wt,
                                                 bf16_t* __restrict__ outb, int D) {
  int lane = threadIdx.x & 31, wave = threadIdx.x >> 5;
  int gr0 = blockIdx.x * 128 + wave * 16;
  int h = blockIdx.y;
  int d0 = blockIdx.z * 64;
  const bf16_t* wth = Wt + (size_t)h * D * INSZ;

  v8f acc[4];
#pragma unroll
  for (int t = 0; t < 4; ++t)
#pragma unroll
    for (int r = 0; r < 8; ++r) acc[t][r] = 0.0f;

  for (int k0 = 0; k0 < INSZ; k0 += 32) {
    v16bf a = load_a_frag(xb + (size_t)gr0 * INSZ + k0, INSZ, lane);
#pragma unroll
    for (int t = 0; t < 4; ++t) {
      v16bf b = load_bT_frag(wth + (size_t)(d0 + t * 16) * INSZ + k0, INSZ, lane);
      acc[t] = wmma_bf16(a, b, acc[t]);
    }
  }
  int half = lane >> 4, n = lane & 15;
#pragma unroll
  for (int t = 0; t < 4; ++t)
#pragma unroll
    for (int r = 0; r < 8; ++r) {
      int gr = gr0 + half * 8 + r;
      outb[((size_t)h * (BATCH * SEQ) + gr) * D + d0 + t * 16 + n] = (bf16_t)acc[t][r];
    }
}

// Flash attention: grid (SEQ/128, BATCH, HEADS); block 256 = 8 waves, 16 q-rows/wave.
__global__ __launch_bounds__(256) void flash_attn(const bf16_t* __restrict__ Qb,
                                                  const bf16_t* __restrict__ Kb,
                                                  const bf16_t* __restrict__ Vb,
                                                  float* __restrict__ out) {
  __shared__ bf16_t kbuf[32 * KD];        // 32 keys x 64 feats (row-major = Bt for QK^T)
  __shared__ bf16_t vtbuf[VD * 32];       // V transposed: [vdim][key] (Bt for P*V)
  __shared__ bf16_t pbuf[8][16 * 32];     // per-wave P tile staging (C-layout -> A-layout)

  int lane = threadIdx.x & 31, wave = threadIdx.x >> 5;
  size_t hb = (size_t)blockIdx.z * BATCH + blockIdx.y;
  const bf16_t* Qhb = Qb + hb * SEQ * KD;
  const bf16_t* Khb = Kb + hb * SEQ * KD;
  const bf16_t* Vhb = Vb + hb * SEQ * VD;
  float* Ohb = out + hb * SEQ * VD;
  int q0 = blockIdx.x * 128 + wave * 16;

  v16bf qf0 = load_a_frag(Qhb + (size_t)q0 * KD + 0,  KD, lane);
  v16bf qf1 = load_a_frag(Qhb + (size_t)q0 * KD + 32, KD, lane);

  v8f acc[8];
  float mrow[8], lrow[8];
#pragma unroll
  for (int t = 0; t < 8; ++t)
#pragma unroll
    for (int r = 0; r < 8; ++r) acc[t][r] = 0.0f;
#pragma unroll
  for (int r = 0; r < 8; ++r) { mrow[r] = -3.0e38f; lrow[r] = 0.0f; }

  const float scale = 0.125f;  // 1/sqrt(64)

  for (int k0 = 0; k0 < SEQ; k0 += 32) {
    __syncthreads();
    // stage K block (contiguous 4 KB): async DMA global->LDS, 16 B per lane.
#if HAVE_ASYNC_LDS
    {
      const bf16_t* g = Khb + (size_t)k0 * KD + threadIdx.x * 8;
      __builtin_amdgcn_global_load_async_to_lds_b128(
          (__attribute__((address_space(1))) v4i*)g,
          (__attribute__((address_space(3))) v4i*)(kbuf + threadIdx.x * 8),
          0, 0);
    }
#else
    {
      const uint32_t* src = (const uint32_t*)(Khb + (size_t)k0 * KD);
      uint32_t* dst = (uint32_t*)kbuf;
      for (int i = threadIdx.x; i < (32 * KD) / 2; i += 256) dst[i] = src[i];
    }
#endif
    // cooperative stage: V block transposed [vdim][key] (transpose needs VGPR path)
    for (int i = threadIdx.x; i < 32 * VD; i += 256) {
      int key = i >> 7, vd = i & (VD - 1);
      vtbuf[vd * 32 + key] = Vhb[(size_t)(k0 + key) * VD + vd];
    }
#if HAVE_ASYNC_LDS
    wait_asynccnt0();
#endif
    __syncthreads();

    // S = Q * K^T : two 16x16 score tiles (keys k0..k0+15 and k0+16..k0+31)
    v8f s0, s1;
#pragma unroll
    for (int r = 0; r < 8; ++r) { s0[r] = 0.0f; s1[r] = 0.0f; }
    s0 = wmma_bf16(qf0, load_bT_frag(kbuf + 0 * KD + 0,  KD, lane), s0);
    s0 = wmma_bf16(qf1, load_bT_frag(kbuf + 0 * KD + 32, KD, lane), s0);
    s1 = wmma_bf16(qf0, load_bT_frag(kbuf + 16 * KD + 0,  KD, lane), s1);
    s1 = wmma_bf16(qf1, load_bT_frag(kbuf + 16 * KD + 32, KD, lane), s1);

    // online softmax: row stats live in 16-lane halves (C-layout rows)
    float corr[8];
#pragma unroll
    for (int r = 0; r < 8; ++r) {
      float a0 = s0[r] * scale, a1 = s1[r] * scale;
      float mx = fmaxf(a0, a1);
      mx = fmaxf(mx, __shfl_xor(mx, 1));
      mx = fmaxf(mx, __shfl_xor(mx, 2));
      mx = fmaxf(mx, __shfl_xor(mx, 4));
      mx = fmaxf(mx, __shfl_xor(mx, 8));
      float mn = fmaxf(mrow[r], mx);
      float c = __expf(mrow[r] - mn);
      mrow[r] = mn;
      corr[r] = c;
      float p0 = __expf(a0 - mn), p1 = __expf(a1 - mn);
      s0[r] = p0; s1[r] = p1;
      float rs = p0 + p1;
      rs += __shfl_xor(rs, 1);
      rs += __shfl_xor(rs, 2);
      rs += __shfl_xor(rs, 4);
      rs += __shfl_xor(rs, 8);
      lrow[r] = lrow[r] * c + rs;
    }
#pragma unroll
    for (int t = 0; t < 8; ++t)
#pragma unroll
      for (int r = 0; r < 8; ++r) acc[t][r] *= corr[r];

    // C-layout -> A-layout via wave-private LDS (same-wave DS ops are in-order)
    bf16_t* pl = pbuf[wave];
    {
      int mb = (lane >> 4) * 8, nn = lane & 15;
#pragma unroll
      for (int r = 0; r < 8; ++r) {
        pl[(mb + r) * 32 + nn]      = (bf16_t)s0[r];
        pl[(mb + r) * 32 + 16 + nn] = (bf16_t)s1[r];
      }
    }
    v16bf pf = load_a_frag(pl, 32, lane);

    // O += P * V : 8 N-tiles of 16 v-dims
#pragma unroll
    for (int t = 0; t < 8; ++t)
      acc[t] = wmma_bf16(pf, load_bT_frag(vtbuf + t * 16 * 32, 32, lane), acc[t]);
  }

  // epilogue: divide by running row sum, store f32 (H,B,S,V)
  int nn = lane & 15, halfq = (lane >> 4) * 8;
#pragma unroll
  for (int r = 0; r < 8; ++r) {
    float inv = 1.0f / lrow[r];
#pragma unroll
    for (int t = 0; t < 8; ++t)
      Ohb[(size_t)(q0 + halfq + r) * VD + t * 16 + nn] = acc[t][r] * inv;
  }
}

extern "C" void kernel_launch(void* const* d_in, const int* in_sizes, int n_in,
                              void* d_out, int out_size, void* d_ws, size_t ws_size,
                              hipStream_t stream) {
  const float* x  = (const float*)d_in[0];
  const float* Mq = (const float*)d_in[1];
  const float* Mk = (const float*)d_in[2];
  const float* Mv = (const float*)d_in[3];
  float* out = (float*)d_out;

  char* ws = (char*)d_ws;
  bf16_t* xb  = (bf16_t*)ws; ws += (size_t)BATCH * SEQ * INSZ * 2;
  bf16_t* Wqt = (bf16_t*)ws; ws += (size_t)HEADS * KD * INSZ * 2;
  bf16_t* Wkt = (bf16_t*)ws; ws += (size_t)HEADS * KD * INSZ * 2;
  bf16_t* Wvt = (bf16_t*)ws; ws += (size_t)HEADS * VD * INSZ * 2;
  bf16_t* Qb  = (bf16_t*)ws; ws += (size_t)HEADS * BATCH * SEQ * KD * 2;
  bf16_t* Kb  = (bf16_t*)ws; ws += (size_t)HEADS * BATCH * SEQ * KD * 2;
  bf16_t* Vb  = (bf16_t*)ws; ws += (size_t)HEADS * BATCH * SEQ * VD * 2;

  (void)in_sizes; (void)n_in; (void)out_size; (void)ws_size;

  int nx = BATCH * SEQ * INSZ;
  cvt_f32_bf16<<<1024, 256, 0, stream>>>(x, xb, nx);
  pack_weight_t<<<512, 256, 0, stream>>>(Mq, Wqt, KD);
  pack_weight_t<<<512, 256, 0, stream>>>(Mk, Wkt, KD);
  pack_weight_t<<<512, 256, 0, stream>>>(Mv, Wvt, VD);

  dim3 gqk((BATCH * SEQ) / 128, HEADS, KD / 64);  // 64,8,1
  proj_gemm<<<gqk, 256, 0, stream>>>(xb, Wqt, Qb, KD);
  proj_gemm<<<gqk, 256, 0, stream>>>(xb, Wkt, Kb, KD);
  dim3 gv((BATCH * SEQ) / 128, HEADS, VD / 64);   // 64,8,2
  proj_gemm<<<gv, 256, 0, stream>>>(xb, Wvt, Vb, VD);

  dim3 ga(SEQ / 128, BATCH, HEADS);               // 16,4,8
  flash_attn<<<ga, 256, 0, stream>>>(Qb, Kb, Vb, out);
}
